// Attention_19378892439641
// MI455X (gfx1250) — matrix operand order repaired and checked by reading, not compiled, not run on hardware
//
#include <hip/hip_runtime.h>

// ---------------- problem constants (match reference) ----------------
#define NB_HEAD 12
#define DH      64
#define DM      768          // NB_HEAD * DH
#define BATCH   8
#define SEQ     1024

// ---------------- vector types for WMMA ----------------
typedef __attribute__((ext_vector_type(16))) __bf16          v16bf;
typedef __attribute__((ext_vector_type(8)))  float           v8f;
typedef __attribute__((ext_vector_type(16))) unsigned short  v16u;
typedef __attribute__((ext_vector_type(8)))  unsigned short  v8u;

union F8 { v8f v; float f[8]; };

__device__ __forceinline__ v8f zerov8() {
    v8f z = {0.f,0.f,0.f,0.f,0.f,0.f,0.f,0.f};
    return z;
}

// f32 -> bf16 using the native convert (lowers to v_cvt*bf16_f32)
__device__ __forceinline__ unsigned short f2bfbits(float f) {
    __bf16 b = (__bf16)f;
    return __builtin_bit_cast(unsigned short, b);
}

__device__ __forceinline__ v8f wmma_bf16(v16bf a, v16bf b, v8f c) {
    // D = A(16x32 bf16) * B(32x16 bf16) + C(16x16 f32)
    return __builtin_amdgcn_wmma_f32_16x16x32_bf16(
        /*neg_a=*/false, a, /*neg_b=*/false, b,
        /*c_mod=*/(short)0, c, /*reuse_a=*/false, /*reuse_b=*/false);
}

// assemble a 16-element fragment vector from two contiguous 8x16-bit chunks
__device__ __forceinline__ v16bf cat8(v8u lo, v8u hi) {
    v16u a = __builtin_shufflevector(lo, hi, 0,1,2,3,4,5,6,7,8,9,10,11,12,13,14,15);
    return __builtin_bit_cast(v16bf, a);
}

// all-ones bf16 B fragment (for row-sum via WMMA)
__device__ __forceinline__ v16bf ones16() {
    v16u u;
    #pragma unroll
    for (int j = 0; j < 16; ++j) u[j] = 0x3F80;   // bf16 1.0
    return __builtin_bit_cast(v16bf, u);
}

// one DPP lane-permute + fmax step (pure VALU, no DS, no waits)
template <int CTRL>
__device__ __forceinline__ float dpp_fmax_step(float x) {
    int v = __builtin_bit_cast(int, x);
    int p = __builtin_amdgcn_update_dpp(v, v, CTRL, 0xF, 0xF, false);
    return fmaxf(x, __builtin_bit_cast(float, p));
}

// max across each 16-lane half-wave via DPP16 row ops:
//   xor1 (quad_perm 1,0,3,2) -> xor2 (quad_perm 2,3,0,1)
//   -> row_half_mirror (crosses the 4-boundary) -> row_mirror (crosses 8)
__device__ __forceinline__ float rowmax16(float x) {
    x = dpp_fmax_step<0x0B1>(x);   // quad_perm [1,0,3,2]
    x = dpp_fmax_step<0x04E>(x);   // quad_perm [2,3,0,1]
    x = dpp_fmax_step<0x141>(x);   // row_half_mirror
    x = dpp_fmax_step<0x140>(x);   // row_mirror
    return x;
}

// =====================================================================
// Converter 0: x (f32, [8192][768]) -> xb (bf16 bits, same layout)
// =====================================================================
__global__ __launch_bounds__(256)
void cvt_x_kernel(const float* __restrict__ in, unsigned short* __restrict__ out)
{
    const size_t i = ((size_t)blockIdx.x * 256 + threadIdx.x) * 8;
    v8f v = *(const v8f*)(in + i);
    v8u o;
    #pragma unroll
    for (int j = 0; j < 8; ++j) o[j] = f2bfbits(v[j]);
    *(v8u*)(out + i) = o;
}

// =====================================================================
// Converter 1: W (f32, [K][N]=[768][768]) -> Wt (bf16 bits, [N][K])
// grid (768, 3), block 256: k = blockIdx.x, n = blockIdx.y*256+tid
// =====================================================================
__global__ __launch_bounds__(256)
void cvt_wt_kernel(const float* __restrict__ W, unsigned short* __restrict__ Wt)
{
    const int k = blockIdx.x;
    const int n = blockIdx.y * 256 + threadIdx.x;
    Wt[(size_t)n * DM + k] = f2bfbits(W[(size_t)k * DM + n]);
}

// =====================================================================
// Phase 1: P = x @ W with bf16 WMMA (all-bf16 operands, f32 accum).
//   VMODE==0 : out[b][h][s][d]   (Q and K)
//   VMODE==1 : out[b][h][d][s]   (V transposed; packed b128 stores)
//   PRESCALE : multiply result by 1/sqrt(dh)=0.125 (exact; used for Q)
// Block = 128 threads (4 waves); wave computes a 16(M) x 64(N) strip.
// Grid = (8192/64, 768/64)
// =====================================================================
template <int VMODE, bool PRESCALE>
__global__ __launch_bounds__(128)
void qkv_gemm_kernel(const unsigned short* __restrict__ xb,
                     const unsigned short* __restrict__ Wt,
                     unsigned short* __restrict__ outb)
{
    const int lane = threadIdx.x & 31;
    const int wave = threadIdx.x >> 5;
    const int l15  = lane & 15;
    const int kh   = lane >> 4;          // which K-half this lane holds
    const int m0   = blockIdx.x * 64 + wave * 16;
    const int n0   = blockIdx.y * 64;

    F8 acc[4];
    #pragma unroll
    for (int t = 0; t < 4; ++t) acc[t].v = zerov8();

    const unsigned short* xrow = xb + (size_t)(m0 + l15) * DM;
    const unsigned short* wrow[4];
    #pragma unroll
    for (int t = 0; t < 4; ++t)
        wrow[t] = Wt + (size_t)(n0 + t*16 + l15) * DM + kh * 16;

    for (int k0 = 0; k0 < DM; k0 += 32) {
        // A fragment: row (m0+l15), K = k0+kh*16 + 0..15 (contiguous)
        v8u lo = *(const v8u*)(xrow + k0 + kh*16);
        v8u hi = *(const v8u*)(xrow + k0 + kh*16 + 8);
        v16bf a = cat8(lo, hi);

        #pragma unroll
        for (int t = 0; t < 4; ++t) {
            // B fragment: col n0+t*16+l15, K = k0+kh*16 + 0..15 (contiguous)
            v16bf b = __builtin_bit_cast(v16bf, *(const v16u*)(wrow[t] + k0));
            acc[t].v = wmma_bf16(a, b, acc[t].v);
        }
    }

    // C layout: reg r -> row (r or r+8), lane&15 -> N
    const int mrow_base = m0 + ((lane < 16) ? 0 : 8);
    const int bidx = mrow_base >> 10;          // / SEQ
    const int s0   = mrow_base & (SEQ - 1);    // % SEQ (rows s0..s0+7, same batch)

    #pragma unroll
    for (int t = 0; t < 4; ++t) {
        const int n  = n0 + t*16 + l15;
        const int h  = n >> 6;                 // / DH
        const int d  = n & (DH - 1);           // % DH
        const int bh = bidx * NB_HEAD + h;
        if (VMODE == 0) {
            #pragma unroll
            for (int r = 0; r < 8; ++r) {
                const float val = PRESCALE ? acc[t].f[r] * 0.125f : acc[t].f[r];
                outb[((size_t)bh * SEQ + (s0 + r)) * DH + d] = f2bfbits(val);
            }
        } else {
            // V^T: rows r are contiguous in s -> one 16-byte store
            v8u pk;
            #pragma unroll
            for (int r = 0; r < 8; ++r) pk[r] = f2bfbits(acc[t].f[r]);
            *(v8u*)(outb + ((size_t)bh * DH + d) * SEQ + s0) = pk;
        }
    }
}

// =====================================================================
// Phase 2: flash attention. One wave = 16 query rows; 4 waves per WG.
// Q pre-scaled by 1/sqrt(dh). Row max via DPP (pure VALU); row sums via
// WMMA (P x ones). DS used only for the P layout conversion.
// Grid = (SEQ/64, NB_HEAD, BATCH), block = 128.
// =====================================================================
__global__ __launch_bounds__(128)
void flash_attn_kernel(const unsigned short* __restrict__ Qb,
                       const unsigned short* __restrict__ Kb,
                       const unsigned short* __restrict__ Vt,
                       float* __restrict__ out)
{
    __shared__ unsigned short lds_p[4][16][32];   // per-wave P staging (bf16)

    const int lane = threadIdx.x & 31;
    const int wave = threadIdx.x >> 5;
    const int l15  = lane & 15;
    const int kh   = lane >> 4;
    const int h    = blockIdx.y;
    const int b    = blockIdx.z;
    const int q0   = blockIdx.x * 64 + wave * 16;

    const unsigned short* Qh = Qb + (size_t)(b*NB_HEAD + h) * SEQ * DH;   // [S][64]
    const unsigned short* Kh = Kb + (size_t)(b*NB_HEAD + h) * SEQ * DH;   // [S][64]
    const unsigned short* Vh = Vt + (size_t)(b*NB_HEAD + h) * DH  * SEQ;  // [64][S]

    // Q A-fragments for the whole pass (dh=64 -> two K=32 chunks)
    v16bf aq[2];
    {
        const unsigned short* qrow = Qh + (size_t)(q0 + l15) * DH;
        #pragma unroll
        for (int c = 0; c < 2; ++c) {
            v8u lo = *(const v8u*)(qrow + c*32 + kh*16);
            v8u hi = *(const v8u*)(qrow + c*32 + kh*16 + 8);
            aq[c] = cat8(lo, hi);
        }
    }

    const v16bf onesb = ones16();

    F8 o[4], ls;          // O accumulators + row-sum accumulator (C layout)
    #pragma unroll
    for (int t = 0; t < 4; ++t) o[t].v = zerov8();
    ls.v = zerov8();
    float mrow[8];
    #pragma unroll
    for (int r = 0; r < 8; ++r) mrow[r] = -__builtin_inff();

    for (int kb = 0; kb < SEQ; kb += 32) {
        // ---- S = Q K^T for 32 keys (two 16-col tiles); Q pre-scaled ----
        F8 s[2];
        #pragma unroll
        for (int t = 0; t < 2; ++t) {
            v8f c = zerov8();
            #pragma unroll
            for (int cc = 0; cc < 2; ++cc) {
                const unsigned short* krow =
                    Kh + (size_t)(kb + t*16 + l15) * DH + cc*32 + kh*16;
                v16bf bk = __builtin_bit_cast(v16bf, *(const v16u*)krow);
                c = wmma_bf16(aq[cc], bk, c);
            }
            s[t].v = c;
        }

        // ---- online softmax: row max via DPP, then exponentiate ----
        #pragma unroll
        for (int r = 0; r < 8; ++r) {
            const float mx   = rowmax16(fmaxf(s[0].f[r], s[1].f[r]));
            const float mnew = fmaxf(mrow[r], mx);
            const float sc   = __expf(mrow[r] - mnew);
            mrow[r] = mnew;
            s[0].f[r] = __expf(s[0].f[r] - mnew);
            s[1].f[r] = __expf(s[1].f[r] - mnew);
            ls.f[r] *= sc;                         // rescale running row-sum
            #pragma unroll
            for (int t = 0; t < 4; ++t) o[t].f[r] *= sc;
        }

        // ---- stage P (C-layout) -> LDS -> reload as A-fragment ----
        // Same-wave DS ops are in-order on CDNA5; region is wave-private.
        const int rbase = (lane < 16) ? 0 : 8;
        #pragma unroll
        for (int r = 0; r < 8; ++r) {
            lds_p[wave][rbase + r][l15]      = f2bfbits(s[0].f[r]);
            lds_p[wave][rbase + r][16 + l15] = f2bfbits(s[1].f[r]);
        }
        v16bf pa;
        {
            const unsigned short* prow = &lds_p[wave][l15][0];
            v8u lo = *(const v8u*)(prow + kh*16);
            v8u hi = *(const v8u*)(prow + kh*16 + 8);
            pa = cat8(lo, hi);
        }

        // ---- row sums via WMMA: ls += P x ones (replicated per lane) ----
        ls.v = wmma_bf16(pa, onesb, ls.v);

        // ---- O += P * V  (4 dh-tiles of 16) ----
        #pragma unroll
        for (int t = 0; t < 4; ++t) {
            const unsigned short* vrow =
                Vh + (size_t)(t*16 + l15) * SEQ + kb + kh*16;
            v16bf bv = __builtin_bit_cast(v16bf, *(const v16u*)vrow);
            o[t].v = wmma_bf16(pa, bv, o[t].v);
        }
    }

    // ---- epilogue: normalize and store f32 output [B][S][H*dh] ----
    const int rbase = (lane < 16) ? 0 : 8;
    #pragma unroll
    for (int r = 0; r < 8; ++r) {
        const float inv = 1.0f / ls.f[r];
        const int srow = q0 + rbase + r;
        float* orow = out + ((size_t)b * SEQ + srow) * DM + h * DH;
        #pragma unroll
        for (int t = 0; t < 4; ++t)
            orow[t*16 + l15] = o[t].f[r] * inv;
    }
}

// =====================================================================
extern "C" void kernel_launch(void* const* d_in, const int* in_sizes, int n_in,
                              void* d_out, int out_size, void* d_ws, size_t ws_size,
                              hipStream_t stream)
{
    (void)in_sizes; (void)n_in; (void)out_size; (void)ws_size;
    const float* x  = (const float*)d_in[0];
    const float* WQ = (const float*)d_in[1];
    const float* WK = (const float*)d_in[2];
    const float* WV = (const float*)d_in[3];
    float* out = (float*)d_out;

    // workspace layout (all bf16 bits as ushort):
    //   xb [8192*768] | WtQ,WtK,WtV [768*768 each] | Q,K,V [B*H*S*dh each]
    const size_t XB  = (size_t)BATCH * SEQ * DM;           // 6,291,456
    const size_t WT  = (size_t)DM * DM;                    //   589,824
    const size_t PER = (size_t)BATCH * NB_HEAD * SEQ * DH; // 6,291,456
    unsigned short* xb   = (unsigned short*)d_ws;
    unsigned short* wtq  = xb  + XB;
    unsigned short* wtk  = wtq + WT;
    unsigned short* wtv  = wtk + WT;
    unsigned short* qbuf = wtv + WT;
    unsigned short* kbuf = qbuf + PER;
    unsigned short* vbuf = kbuf + PER;

    // converters
    cvt_x_kernel<<<dim3(XB / (256 * 8)), 256, 0, stream>>>(x, xb);
    dim3 gw(DM, DM / 256);
    cvt_wt_kernel<<<gw, 256, 0, stream>>>(WQ, wtq);
    cvt_wt_kernel<<<gw, 256, 0, stream>>>(WK, wtk);
    cvt_wt_kernel<<<gw, 256, 0, stream>>>(WV, wtv);

    // QKV projections (bf16 WMMA); Q pre-scaled by 1/sqrt(dh)
    dim3 g1((BATCH * SEQ) / 64, DM / 64);   // (128, 12)
    qkv_gemm_kernel<0, true ><<<g1, 128, 0, stream>>>(xb, wtq, qbuf);
    qkv_gemm_kernel<0, false><<<g1, 128, 0, stream>>>(xb, wtk, kbuf);
    qkv_gemm_kernel<1, false><<<g1, 128, 0, stream>>>(xb, wtv, vbuf);

    // fused softmax(QK^T)V
    dim3 g2(SEQ / 64, NB_HEAD, BATCH);
    flash_attn_kernel<<<g2, 128, 0, stream>>>(qbuf, kbuf, vbuf, out);
}
